// LMRayMarcher_33861522161820
// MI455X (gfx1250) — matrix operand order repaired and checked by reading, not compiled, not run on hardware
//
#include <hip/hip_runtime.h>
#include <hip/hip_bf16.h>

typedef __attribute__((ext_vector_type(16))) _Float16 v16h;
typedef __attribute__((ext_vector_type(8)))  _Float16 v8h;
typedef __attribute__((ext_vector_type(8)))  float    v8f;

#define HIDDEN          128
#define STRIDE          136          // padded f16 row stride (272B, 16B-aligned, bank-conflict-free)
#define RAYS_PER_WAVE   16
#define WAVES_PER_BLOCK 8
#define THREADS         (WAVES_PER_BLOCK * 32)
#define RAYS_PER_BLOCK  (WAVES_PER_BLOCK * RAYS_PER_WAVE)
#define LM_LAMB         1e-3f
#define LM_ITERS        20

// ---- LDS layout (byte offsets) ----
#define OFF_W2   0                    // 128x128 f16, swizzled to B-fragment layout (32768 B)
#define OFF_W3   32768                // 32768 B
#define OFF_W1   65536                // 3x128 f32   (1536 B)
#define OFF_B1   67072                // 128 f32     (512 B)
#define OFF_B2   67584
#define OFF_B3   68096
#define OFF_W4   68608                // 128 f32
#define OFF_B4   69120                // 1 f32 (+pad)
#define OFF_WAVE 69136
// per-wave region
#define W_HP0    0                    // 16 x STRIDE f16 (4352 B) primal act, buffer 0
#define W_HT0    4352                 // tangent act, buffer 0
#define W_HP1    8704                 // buffer 1
#define W_HT1    13056
#define W_PX     17408                // 16x3 f32 ray origins
#define W_PU     17600                // 16x3 f32 ray dirs
#define W_D      17792                // 16 f32 current depths
#define W_FJ     17856                // 32 f32 {f[0..15], J[0..15]}
#define WAVE_SZ  17984
#define LDS_TOTAL (OFF_WAVE + WAVES_PER_BLOCK * WAVE_SZ)   // 213008 B < 320KB/WGP

// Assemble a 16x32 f16 A-fragment for one K-block from the padded activation buffer.
// 16-bit A layout: lanes 0-15 hold K=colsel..colsel+7 (v0-3) and +16 (v4-7); lanes 16-31 use colsel=8.
__device__ __forceinline__ v16h ldsA(const char* buf, int row, int col) {
  const v8h* lo = (const v8h*)(buf + (row * STRIDE + col) * 2);
  const v8h* hi = (const v8h*)(buf + (row * STRIDE + col + 16) * 2);
  return __builtin_shufflevector(*lo, *hi, 0,1,2,3,4,5,6,7,8,9,10,11,12,13,14,15);
}

// One dense 128->128 layer (primal + JVP tangent) for 16 rays on one wave.
// inOff/outOff point at {hp, ht} pairs (ht = hp + 4352).
__device__ __forceinline__ void dense_layer(char* sm, int inOff, int outOff,
                                            int wswOff, int biasOff, int lane)
{
  const char* inP = sm + inOff;
  const char* inT = sm + inOff + 4352;
  _Float16*   outP = (_Float16*)(sm + outOff);
  _Float16*   outT = (_Float16*)(sm + outOff + 4352);
  const _Float16* Wsw = (const _Float16*)(sm + wswOff);
  const float*    bias = (const float*)(sm + biasOff);

  const int r      = lane & 15;          // A-row this lane reads
  const int colsel = (lane >> 4) << 3;   // 0 or 8: K sub-offset per A layout
  const int rowadd = (lane >> 4) << 3;   // C/D: lanes 16-31 hold rows M+8
  const int n0     = lane & 15;          // C/D: output feature column

#pragma unroll
  for (int nt = 0; nt < 8; ++nt) {       // 8 output feature tiles of 16
    float bv = bias[nt * 16 + n0];
    v8f accp, acct;
#pragma unroll
    for (int i = 0; i < 8; ++i) { accp[i] = bv; acct[i] = 0.f; }

#pragma unroll
    for (int kb = 0; kb < 4; ++kb) {     // K = 4 x 32
      v16h ap = ldsA(inP, r, kb * 32 + colsel);
      v16h at = ldsA(inT, r, kb * 32 + colsel);
      // pre-swizzled B fragment: 32 contiguous bytes per lane
      v16h bm = *(const v16h*)(Wsw + (((kb * 8 + nt) * 32 + lane) << 4));
      accp = __builtin_amdgcn_wmma_f32_16x16x32_f16(false, ap, false, bm,
                                                    (short)0, accp, false, false);
      acct = __builtin_amdgcn_wmma_f32_16x16x32_f16(false, at, false, bm,
                                                    (short)0, acct, false, false);
    }
    // ReLU on primal, ReLU-mask on tangent, stage as f16 for next layer's A
#pragma unroll
    for (int i = 0; i < 8; ++i) {
      int ray = i + rowadd;
      float z = accp[i], zt = acct[i];
      outP[ray * STRIDE + nt * 16 + n0] = (_Float16)fmaxf(z, 0.f);
      outT[ray * STRIDE + nt * 16 + n0] = (_Float16)(z > 0.f ? zt : 0.f);
    }
  }
}

__global__ void __launch_bounds__(THREADS)
lm_raymarch_kernel(const float* __restrict__ gx, const float* __restrict__ gu,
                   const float* __restrict__ gW1, const float* __restrict__ gb1,
                   const float* __restrict__ gW2, const float* __restrict__ gb2,
                   const float* __restrict__ gW3, const float* __restrict__ gb3,
                   const float* __restrict__ gW4, const float* __restrict__ gb4,
                   float* __restrict__ out)
{
  extern __shared__ __align__(32) char smem[];
  const int tid  = threadIdx.x;
  const int lane = tid & 31;
  const int wave = tid >> 5;
  const int rayBase = blockIdx.x * RAYS_PER_BLOCK + wave * RAYS_PER_WAVE;
  char* wsm = smem + OFF_WAVE + wave * WAVE_SZ;

  // ---------------- one-time setup ----------------
  {
    // Swizzle W2/W3 (f32 global, [in][out] row-major) into f16 B-fragment layout:
    // fragment (kb,nt), lane ln, vgpr v  holds  W[K0+2v+(v>=4?8:0)][N], W[K0+2v+1+(v>=4?8:0)][N]
    const float* gw[2] = { gW2, gW3 };
#pragma unroll
    for (int m = 0; m < 2; ++m) {
      _Float16* dstW = (_Float16*)(smem + (m ? OFF_W3 : OFF_W2));
      for (int item = tid; item < 1024; item += THREADS) {   // 32 frags x 32 lanes
        int frag = item >> 5, ln = item & 31;
        int kb = frag >> 3, nt = frag & 7;
        int K0 = kb * 32 + ((ln >> 4) << 3);
        int N  = nt * 16 + (ln & 15);
        _Float16* dst = dstW + (item << 4);
#pragma unroll
        for (int v = 0; v < 8; ++v) {
          dst[2 * v]     = (_Float16)gw[m][(K0 + 2 * v + (v >= 4 ? 8 : 0))     * HIDDEN + N];
          dst[2 * v + 1] = (_Float16)gw[m][(K0 + 2 * v + 1 + (v >= 4 ? 8 : 0)) * HIDDEN + N];
        }
      }
    }
    float* w1 = (float*)(smem + OFF_W1);
    for (int i = tid; i < 3 * HIDDEN; i += THREADS) w1[i] = gW1[i];
    float* b1 = (float*)(smem + OFF_B1);
    float* b2 = (float*)(smem + OFF_B2);
    float* b3 = (float*)(smem + OFF_B3);
    float* w4 = (float*)(smem + OFF_W4);
    for (int i = tid; i < HIDDEN; i += THREADS) {
      b1[i] = gb1[i]; b2[i] = gb2[i]; b3[i] = gb3[i]; w4[i] = gW4[i];
    }
    if (tid == 0) *(float*)(smem + OFF_B4) = gb4[0];

    float* px = (float*)(wsm + W_PX);
    float* pu = (float*)(wsm + W_PU);
    for (int i = lane; i < RAYS_PER_WAVE * 3; i += 32) {
      px[i] = gx[rayBase * 3 + i];
      pu[i] = gu[rayBase * 3 + i];
    }
    float* dv = (float*)(wsm + W_D);
    if (lane < 16) dv[lane] = 0.f;
  }
  __syncthreads();   // weights visible to all waves; everything after is intra-wave

  const float  b4v = *(const float*)(smem + OFF_B4);
  const float* W1f = (const float*)(smem + OFF_W1);
  const float* B1f = (const float*)(smem + OFF_B1);
  const float* W4f = (const float*)(smem + OFF_W4);
  float* px = (float*)(wsm + W_PX);
  float* pu = (float*)(wsm + W_PU);
  float* dv = (float*)(wsm + W_D);
  float* fj = (float*)(wsm + W_FJ);

  for (int it = 0; it < LM_ITERS; ++it) {
    // ---- layer 1: p = x + d*u; z = p@W1+b1 ; zt = u@W1 (tiny, VALU) ----
    {
      _Float16* hp = (_Float16*)(wsm + W_HP0);
      _Float16* ht = (_Float16*)(wsm + W_HT0);
      for (int idx = lane; idx < RAYS_PER_WAVE * HIDDEN; idx += 32) {
        int r = idx >> 7, f = idx & 127;
        float dd = dv[r];
        float z = B1f[f], zt = 0.f;
#pragma unroll
        for (int c = 0; c < 3; ++c) {
          float w  = W1f[c * HIDDEN + f];
          float uc = pu[r * 3 + c];
          z  = fmaf(fmaf(dd, uc, px[r * 3 + c]), w, z);
          zt = fmaf(uc, w, zt);
        }
        hp[r * STRIDE + f] = (_Float16)fmaxf(z, 0.f);
        ht[r * STRIDE + f] = (_Float16)(z > 0.f ? zt : 0.f);
      }
    }
    // ---- layers 2 & 3: WMMA, ping-pong buf0 -> buf1 -> buf0 ----
    dense_layer(smem, OFF_WAVE + wave * WAVE_SZ + W_HP0,
                      OFF_WAVE + wave * WAVE_SZ + W_HP1, OFF_W2, OFF_B2, lane);
    dense_layer(smem, OFF_WAVE + wave * WAVE_SZ + W_HP1,
                      OFF_WAVE + wave * WAVE_SZ + W_HP0, OFF_W3, OFF_B3, lane);
    // ---- layer 4 (128->1) + LM update: lanes 0-15 compute f, 16-31 compute J ----
    {
      int r = lane & 15;
      const char* src = wsm + ((lane < 16) ? W_HP0 : W_HT0);
      float acc = 0.f;
#pragma unroll
      for (int k8 = 0; k8 < 16; ++k8) {
        v8h hv = *(const v8h*)(src + (r * STRIDE + k8 * 8) * 2);
#pragma unroll
        for (int j = 0; j < 8; ++j) acc = fmaf((float)hv[j], W4f[k8 * 8 + j], acc);
      }
      if (lane < 16) acc += b4v;
      fj[lane] = acc;                 // DS ops are in-order within a wave
      if (lane < 16) {
        float f = fj[r], J = fj[16 + r];
        dv[r] = dv[r] - J * f / (J * J + LM_LAMB);
      }
    }
  }

  if (lane < 16) out[rayBase + lane] = dv[lane];
}

extern "C" void kernel_launch(void* const* d_in, const int* in_sizes, int n_in,
                              void* d_out, int out_size, void* d_ws, size_t ws_size,
                              hipStream_t stream) {
  const float* x  = (const float*)d_in[0];
  const float* u  = (const float*)d_in[1];
  const float* W1 = (const float*)d_in[2];
  const float* b1 = (const float*)d_in[3];
  const float* W2 = (const float*)d_in[4];
  const float* b2 = (const float*)d_in[5];
  const float* W3 = (const float*)d_in[6];
  const float* b3 = (const float*)d_in[7];
  const float* W4 = (const float*)d_in[8];
  const float* b4 = (const float*)d_in[9];
  float* out = (float*)d_out;

  int nRays  = in_sizes[0] / 3;
  int blocks = nRays / RAYS_PER_BLOCK;

  lm_raymarch_kernel<<<blocks, THREADS, LDS_TOTAL, stream>>>(
      x, u, W1, b1, W2, b2, W3, b3, W4, b4, out);
}
